// NoRefRetIQANet_79946521247798
// MI455X (gfx1250) — matrix operand order repaired and checked by reading, not compiled, not run on hardware
//
#include <hip/hip_runtime.h>
#include <hip/hip_bf16.h>

typedef __attribute__((ext_vector_type(2)))  float  v2f;
typedef __attribute__((ext_vector_type(8)))  float  v8f;
typedef __attribute__((ext_vector_type(8)))  __bf16 v8bf;   // 16-byte chunk
typedef __attribute__((ext_vector_type(16))) __bf16 v16bf;  // WMMA A/B operand

// Tile configuration
constexpr int MQ   = 256;  // queries (M dim) == B
constexpr int NTI  = 64;   // bank rows per workgroup tile (N dim)
constexpr int KT   = 32;   // K chunk per stage == bf16 WMMA K
constexpr int LDB  = 40;   // padded LDS row stride in bf16 elems (80 B)
constexpr int TOPK = 9;

static __device__ __forceinline__ v8f wmma_bf16(v16bf a, v16bf b, v8f c) {
  // D = A(16x32 bf16) * B(32x16 bf16) + C(16x16 f32)
  return __builtin_amdgcn_wmma_f32_16x16x32_bf16(false, a, false, b, (short)0,
                                                 c, false, false);
}

// assemble a 16-elem bf16 fragment from two 16-byte LDS chunks
static __device__ __forceinline__ v16bf frag16(const __bf16* c0p,
                                               const __bf16* c1p) {
  v8bf c0 = *(const v8bf*)c0p;
  v8bf c1 = *(const v8bf*)c1p;
  v16bf r;
#pragma unroll
  for (int j = 0; j < 8; ++j) {
    r[j] = c0[j];
    r[8 + j] = c1[j];
  }
  return r;
}

static __device__ __forceinline__ void split_bf16(float x, __bf16& h,
                                                  __bf16& l) {
  h = (__bf16)x;
  l = (__bf16)(x - (float)h);
}

// One-time query split: f32 -> (hi, lo) bf16. 6 MB total, stays in L2.
__global__ __launch_bounds__(256) void split_query_kernel(
    const float* __restrict__ Q, __bf16* __restrict__ Qhi,
    __bf16* __restrict__ Qlo, int total) {
  int i = blockIdx.x * 256 + threadIdx.x;
  if (i < total) {
    __bf16 h, l;
    split_bf16(Q[i], h, l);
    Qhi[i] = h;
    Qlo[i] = l;
  }
}

// scores[m][n] = (Q[m,:].Bank[n,:]) * rsqrt(||Bank[n,:]||^2)
// dot emulated in fp32-accuracy via split-bf16: hi*hi + hi*lo + lo*hi.
// Query norms are per-row constants -> irrelevant for row-wise top-K.
__global__ __launch_bounds__(256) void gemm_norm_kernel(
    const __bf16* __restrict__ Qhi,  // [MQ, D]
    const __bf16* __restrict__ Qlo,  // [MQ, D]
    const float* __restrict__ Bank,  // [N,  D] f32 (split in-register)
    float* __restrict__ S,           // [MQ, N]
    int D, int N) {
  __shared__ __bf16 A_hi[MQ * LDB];   // 20 KB
  __shared__ __bf16 A_lo[MQ * LDB];   // 20 KB
  __shared__ __bf16 B_hi[NTI * LDB];  //  5 KB
  __shared__ __bf16 B_lo[NTI * LDB];  //  5 KB
  __shared__ float ssq_lds[256];
  __shared__ float rnorm_lds[NTI];

  const int t    = threadIdx.x;
  const int lane = t & 31;
  const int wv   = t >> 5;     // wave 0..7 (wave32)
  const int half = lane >> 4;  // 0: lanes 0-15, 1: lanes 16-31
  const int lq   = lane & 15;

  const int n0 = blockIdx.x * NTI;

  // staging map: bank -> 4 threads/row, 8 f32 each; queries -> row t
  const int br = t >> 2;
  const int bc = (t & 3) * 8;
  int bank_row = n0 + br;
  if (bank_row >= N) bank_row = N - 1;  // clamp; stores guarded below
  const float* bank_ptr = Bank + (size_t)bank_row * D;
  const __bf16* qh_ptr = Qhi + (size_t)t * D;
  const __bf16* ql_ptr = Qlo + (size_t)t * D;

  v8f acc[2][4];
#pragma unroll
  for (int i = 0; i < 2; ++i)
#pragma unroll
    for (int j = 0; j < 4; ++j) {
      v8f z = {0.f, 0.f, 0.f, 0.f, 0.f, 0.f, 0.f, 0.f};
      acc[i][j] = z;
    }

  float ssq = 0.f;
  v8bf ahr[4], alr[4];  // query slab: 4x16B hi + 4x16B lo per stage
  float4 bregs[2];      // bank slab: 8 f32 per thread per stage

  // prologue: stage 0 into registers
#pragma unroll
  for (int i = 0; i < 4; ++i) {
    ahr[i] = *(const v8bf*)(qh_ptr + i * 8);
    alr[i] = *(const v8bf*)(ql_ptr + i * 8);
  }
  bregs[0] = *(const float4*)(bank_ptr + bc);
  bregs[1] = *(const float4*)(bank_ptr + bc + 4);

  const int nStages = D / KT;
  for (int st = 0; st < nStages; ++st) {
    __syncthreads();  // prior stage's LDS reads done
    // queries: regs -> LDS (already bf16)
#pragma unroll
    for (int i = 0; i < 4; ++i) {
      *(v8bf*)&A_hi[t * LDB + i * 8] = ahr[i];
      *(v8bf*)&A_lo[t * LDB + i * 8] = alr[i];
    }
    // bank: split f32 -> hi/lo in registers, accumulate sum-of-squares
    {
      float xs[8] = {bregs[0].x, bregs[0].y, bregs[0].z, bregs[0].w,
                     bregs[1].x, bregs[1].y, bregs[1].z, bregs[1].w};
      v8bf vh, vl;
#pragma unroll
      for (int i = 0; i < 8; ++i) {
        __bf16 h, l;
        split_bf16(xs[i], h, l);
        vh[i] = h;
        vl[i] = l;
        ssq += xs[i] * xs[i];
      }
      *(v8bf*)&B_hi[br * LDB + bc] = vh;
      *(v8bf*)&B_lo[br * LDB + bc] = vl;
    }
    __syncthreads();  // slab visible

    // prefetch next stage into registers (hidden under WMMA)
    if (st + 1 < nStages) {
      const size_t ko = (size_t)(st + 1) * KT;
#pragma unroll
      for (int i = 0; i < 4; ++i) {
        ahr[i] = *(const v8bf*)(qh_ptr + ko + i * 8);
        alr[i] = *(const v8bf*)(ql_ptr + ko + i * 8);
      }
      bregs[0] = *(const float4*)(bank_ptr + ko + bc);
      bregs[1] = *(const float4*)(bank_ptr + ko + bc + 4);
      if (st + 2 < nStages)
        __builtin_prefetch(bank_ptr + (size_t)(st + 2) * KT + bc, 0, 1);
    }

    // fragments (ISA 7.12.2):
    //  A 16x32: lane=row M, VGPR0-3 = K 8h+0..7, VGPR4-7 = K 16+8h+0..7
    //  B 32x16: lane=col N, K 8h+0..7, 16+8h+0..7 (same as A)
    const int ar0 = (wv * 32 + lq) * LDB;
    const int ar1 = (wv * 32 + 16 + lq) * LDB;
    v16bf a0h = frag16(&A_hi[ar0 + 8 * half], &A_hi[ar0 + 16 + 8 * half]);
    v16bf a0l = frag16(&A_lo[ar0 + 8 * half], &A_lo[ar0 + 16 + 8 * half]);
    v16bf a1h = frag16(&A_hi[ar1 + 8 * half], &A_hi[ar1 + 16 + 8 * half]);
    v16bf a1l = frag16(&A_lo[ar1 + 8 * half], &A_lo[ar1 + 16 + 8 * half]);

    v16bf bh[4];
#pragma unroll
    for (int ns = 0; ns < 4; ++ns) {
      const int brow = (ns * 16 + lq) * LDB + 8 * half;
      bh[ns] = frag16(&B_hi[brow], &B_hi[brow + 16]);
    }
    // pass 1: hi*hi  (acc reuse distance 8 -> no WMMA RAW stalls)
#pragma unroll
    for (int ns = 0; ns < 4; ++ns) acc[0][ns] = wmma_bf16(a0h, bh[ns], acc[0][ns]);
#pragma unroll
    for (int ns = 0; ns < 4; ++ns) acc[1][ns] = wmma_bf16(a1h, bh[ns], acc[1][ns]);
    // pass 2: lo*hi
#pragma unroll
    for (int ns = 0; ns < 4; ++ns) acc[0][ns] = wmma_bf16(a0l, bh[ns], acc[0][ns]);
#pragma unroll
    for (int ns = 0; ns < 4; ++ns) acc[1][ns] = wmma_bf16(a1l, bh[ns], acc[1][ns]);
    // pass 3: hi*lo
    v16bf bl[4];
#pragma unroll
    for (int ns = 0; ns < 4; ++ns) {
      const int brow = (ns * 16 + lq) * LDB + 8 * half;
      bl[ns] = frag16(&B_lo[brow], &B_lo[brow + 16]);
    }
#pragma unroll
    for (int ns = 0; ns < 4; ++ns) acc[0][ns] = wmma_bf16(a0h, bl[ns], acc[0][ns]);
#pragma unroll
    for (int ns = 0; ns < 4; ++ns) acc[1][ns] = wmma_bf16(a1h, bl[ns], acc[1][ns]);
  }

  // reduce bank sum-of-squares: 4 partials per row
  ssq_lds[t] = ssq;
  __syncthreads();
  if (t < NTI) {
    float s = ssq_lds[4 * t] + ssq_lds[4 * t + 1] + ssq_lds[4 * t + 2] +
              ssq_lds[4 * t + 3];
    rnorm_lds[t] = rsqrtf(s);
  }
  __syncthreads();

  // C layout: VGPR j, lane L -> M = j + 8*(L/16), N = L%16
#pragma unroll
  for (int ms = 0; ms < 2; ++ms) {
#pragma unroll
    for (int ns = 0; ns < 4; ++ns) {
      const int n = n0 + ns * 16 + lq;
      const float rn = rnorm_lds[ns * 16 + lq];
      if (n < N) {
        const int mbase = wv * 32 + ms * 16 + half * 8;
#pragma unroll
        for (int j = 0; j < 8; ++j)
          S[(size_t)(mbase + j) * N + n] = acc[ms][ns][j] * rn;
      }
    }
  }
}

// Per query row: top-9 of sem scores + top-9 of dst scores, mean of metrics.
__global__ __launch_bounds__(256) void topk_mean_kernel(
    const float* __restrict__ Ssem, const float* __restrict__ Sdst,
    const float* __restrict__ metrics, float* __restrict__ out, int N) {
  __shared__ float cv[256 * TOPK];
  __shared__ int ci[256 * TOPK];
  __shared__ float cv2[32 * TOPK];
  __shared__ int ci2[32 * TOPK];

  const int b = blockIdx.x;
  const int t = threadIdx.x;
  const float NEG = -3.402823e38f;
  float total = 0.f;  // thread 0

  for (int branch = 0; branch < 2; ++branch) {
    const float* row = (branch == 0 ? Ssem : Sdst) + (size_t)b * N;

    // phase 1: per-thread sorted top-9 over strided slice
    float v[TOPK];
    int id[TOPK];
#pragma unroll
    for (int j = 0; j < TOPK; ++j) { v[j] = NEG; id[j] = 0; }
    for (int n = t; n < N; n += 256) {
      float s = row[n];
      if (s > v[TOPK - 1]) {
        v[TOPK - 1] = s;
        id[TOPK - 1] = n;
#pragma unroll
        for (int j = TOPK - 1; j > 0; --j) {
          if (v[j] > v[j - 1]) {
            float tv = v[j]; v[j] = v[j - 1]; v[j - 1] = tv;
            int ti = id[j]; id[j] = id[j - 1]; id[j - 1] = ti;
          }
        }
      }
    }
#pragma unroll
    for (int j = 0; j < TOPK; ++j) {
      cv[t * TOPK + j] = v[j];
      ci[t * TOPK + j] = id[j];
    }
    __syncthreads();

    // phase 2: wave 0 reduces 2304 candidates -> 32x9
    if (t < 32) {
      float w[TOPK];
      int wid[TOPK];
#pragma unroll
      for (int j = 0; j < TOPK; ++j) { w[j] = NEG; wid[j] = 0; }
      for (int e = t; e < 256 * TOPK; e += 32) {
        float s = cv[e];
        if (s > w[TOPK - 1]) {
          w[TOPK - 1] = s;
          wid[TOPK - 1] = ci[e];
#pragma unroll
          for (int j = TOPK - 1; j > 0; --j) {
            if (w[j] > w[j - 1]) {
              float tv = w[j]; w[j] = w[j - 1]; w[j - 1] = tv;
              int ti = wid[j]; wid[j] = wid[j - 1]; wid[j - 1] = ti;
            }
          }
        }
      }
#pragma unroll
      for (int j = 0; j < TOPK; ++j) {
        cv2[t * TOPK + j] = w[j];
        ci2[t * TOPK + j] = wid[j];
      }
    }
    __syncthreads();

    // phase 3: thread 0 extracts global top-9, gathers metrics
    if (t == 0) {
      float sum = 0.f;
      for (int k = 0; k < TOPK; ++k) {
        float best = NEG;
        int bp = 0;
        for (int e = 0; e < 32 * TOPK; ++e) {
          if (cv2[e] > best) { best = cv2[e]; bp = e; }
        }
        sum += metrics[ci2[bp]];
        cv2[bp] = NEG;
      }
      total += sum;
    }
    __syncthreads();
  }
  if (t == 0) out[b] = total * (1.0f / (2.0f * TOPK));
}

extern "C" void kernel_launch(void* const* d_in, const int* in_sizes, int n_in,
                              void* d_out, int out_size, void* d_ws,
                              size_t ws_size, hipStream_t stream) {
  const float* f_content = (const float*)d_in[0];  // [256, 4096]
  const float* f_dist    = (const float*)d_in[1];  // [256, 2048]
  const float* sem_bank  = (const float*)d_in[2];  // [N, 4096]
  const float* dst_bank  = (const float*)d_in[3];  // [N, 2048]
  const float* metrics   = (const float*)d_in[4];  // [N]
  // d_in[5] = K (always 9; compile-time TOPK)

  const int N  = in_sizes[4];
  const int DS = in_sizes[2] / N;
  const int DD = in_sizes[3] / N;

  // workspace layout
  float* s_sem = (float*)d_ws;               // [256, N] f32
  float* s_dst = s_sem + (size_t)MQ * N;     // [256, N] f32
  __bf16* qb   = (__bf16*)(s_dst + (size_t)MQ * N);
  __bf16* qhi_sem = qb;
  __bf16* qlo_sem = qhi_sem + (size_t)MQ * DS;
  __bf16* qhi_dst = qlo_sem + (size_t)MQ * DS;
  __bf16* qlo_dst = qhi_dst + (size_t)MQ * DD;

  const int totS = MQ * DS, totD = MQ * DD;
  split_query_kernel<<<(totS + 255) / 256, 256, 0, stream>>>(f_content, qhi_sem,
                                                             qlo_sem, totS);
  split_query_kernel<<<(totD + 255) / 256, 256, 0, stream>>>(f_dist, qhi_dst,
                                                             qlo_dst, totD);

  const int tiles = (N + NTI - 1) / NTI;
  gemm_norm_kernel<<<tiles, 256, 0, stream>>>(qhi_sem, qlo_sem, sem_bank, s_sem,
                                              DS, N);
  gemm_norm_kernel<<<tiles, 256, 0, stream>>>(qhi_dst, qlo_dst, dst_bank, s_dst,
                                              DD, N);
  topk_mean_kernel<<<MQ, 256, 0, stream>>>(s_sem, s_dst, metrics, (float*)d_out,
                                           N);
}
